// LevySDEModel_79164837200659
// MI455X (gfx1250) — matrix operand order repaired and checked by reading, not compiled, not run on hardware
//
#include <hip/hip_runtime.h>
#include <stdint.h>
#include <stddef.h>

// Problem constants (from reference): B,T,D,H,OUT,NLAYERS = 1024,200,64,512,64,3
#define B_    1024
#define T_    200
#define D_    64
#define H_    512
#define OUT_  64
#define T1_   199          // T-1 steps
#define C4_   256          // coeffs last dim = 4*D
#define ZRAW_ 577          // 1 + H + D
#define ZK_   608          // ZRAW padded up to multiple of 32

typedef __attribute__((ext_vector_type(16))) __bf16 v16bf;
typedef __attribute__((ext_vector_type(8)))  __bf16 bf16x8;
typedef __attribute__((ext_vector_type(8)))  float  v8f;

__device__ __forceinline__ uint16_t f2bf(float f) {
  uint32_t u = __float_as_uint(f);
  u += 0x7FFFu + ((u >> 16) & 1u);     // round-to-nearest-even
  return (uint16_t)(u >> 16);
}

// ---- WMMA fragment loads -------------------------------------------------
// A: 16x32 (MxK) bf16, row-major source, leading dim lda.
// Lane L: row M = L&15. Lanes 0-15 hold K {0..7,16..23}; lanes 16-31 hold
// K {8..15,24..31} (ISA 7.12.2 16-bit A layout). Two b128 loads per lane.
__device__ __forceinline__ v16bf load_a_frag(const uint16_t* __restrict__ A,
                                             int lda, int row0, int kk, int lane) {
  const int r    = row0 + (lane & 15);
  const int koff = (lane >> 4) << 3;                 // 0 or 8
  const uint16_t* p = A + (size_t)r * lda + kk + koff;
  v16bf out;
  ((bf16x8*)&out)[0] = *(const bf16x8*)(p);          // K = kk+koff   .. +7
  ((bf16x8*)&out)[1] = *(const bf16x8*)(p + 16);     // K = kk+16+koff.. +7
  return out;
}

// B: 32x16 (KxN) bf16; weights row-major (N,K): lane L -> column N = L&15,
// K {0..7,16..23} for lanes 0-15, K {8..15,24..31} for lanes 16-31 (same as A).
__device__ __forceinline__ v16bf load_b_frag(const uint16_t* __restrict__ W,
                                             int ldw, int col0, int kk, int lane) {
  const int c    = col0 + (lane & 15);
  const int koff = (lane >> 4) << 3;                 // 0 or 8
  const uint16_t* p = W + (size_t)c * ldw + kk + koff;
  v16bf out;
  ((bf16x8*)&out)[0] = *(const bf16x8*)(p);
  ((bf16x8*)&out)[1] = *(const bf16x8*)(p + 16);
  return out;
}

// ---- Dual-path fused GEMM: C = A * W^T + bias, epilogue options ----------
// blockIdx.z selects path (0=f/drift, 1=g/jump): path 1 shifts A columns by
// colOffA, selects W1/bias1, and shifts C columns by colOffC. Launch with
// gridDim.z=1 for single-path GEMMs (W1==W0 etc).
// mode 0: bias            -> bf16 Cb
// mode 1: bias + lipswish -> bf16 Cb
// mode 2: bias            -> f32  Cf
// mode 3: bias            -> f32 Cf AND bf16 Cb
// Block: 128 threads (4 wave32), tile 64(M) x 64(N); wave tile 32x32.
// Grid  (M/64, N/64, paths); all dims divide exactly at every call site.
__global__ __launch_bounds__(128)
void gemm_bf16_wmma(const uint16_t* __restrict__ Abase, int lda,
                    const uint16_t* __restrict__ W0, const uint16_t* __restrict__ W1,
                    const float* __restrict__ bias0, const float* __restrict__ bias1,
                    float* __restrict__ Cf, int ldc,
                    uint16_t* __restrict__ Cb, int ldcb,
                    int K, int mode, int colOffA, int colOffC) {
  const int path = blockIdx.z;
  const uint16_t* __restrict__ A    = Abase + (path ? colOffA : 0);
  const uint16_t* __restrict__ W    = path ? W1   : W0;
  const float*    __restrict__ bias = path ? bias1 : bias0;
  const int cOff = path ? colOffC : 0;

  const int lane = threadIdx.x & 31;
  const int wv   = threadIdx.x >> 5;
  const int m0   = blockIdx.x * 64 + (wv & 1) * 32;
  const int n0   = blockIdx.y * 64 + (wv >> 1) * 32;

  v8f acc00 = {}, acc01 = {}, acc10 = {}, acc11 = {};
  for (int kk = 0; kk < K; kk += 32) {
    v16bf a0 = load_a_frag(A, lda, m0,      kk, lane);
    v16bf a1 = load_a_frag(A, lda, m0 + 16, kk, lane);
    v16bf b0 = load_b_frag(W, K,   n0,      kk, lane);
    v16bf b1 = load_b_frag(W, K,   n0 + 16, kk, lane);
    acc00 = __builtin_amdgcn_wmma_f32_16x16x32_bf16(false, a0, false, b0, (short)0, acc00, false, false);
    acc01 = __builtin_amdgcn_wmma_f32_16x16x32_bf16(false, a0, false, b1, (short)0, acc01, false, false);
    acc10 = __builtin_amdgcn_wmma_f32_16x16x32_bf16(false, a1, false, b0, (short)0, acc10, false, false);
    acc11 = __builtin_amdgcn_wmma_f32_16x16x32_bf16(false, a1, false, b1, (short)0, acc11, false, false);
  }

  // C/D layout: VGPR r -> M = r + (lane>=16 ? 8 : 0); N = lane&15.
  const int mlo = (lane >> 4) * 8;
  const int nin = lane & 15;
  auto epi = [&](v8f acc, int mt, int nt) {
    const int n  = nt + nin;            // column within this path (bias index)
    const int nc = n + cOff;            // column in the (possibly paired) C
    const float bn = bias[n];
    #pragma unroll
    for (int r = 0; r < 8; ++r) {
      const int m = mt + mlo + r;
      float v = acc[r] + bn;
      if (mode == 0) {
        Cb[(size_t)m * ldcb + nc] = f2bf(v);
      } else if (mode == 1) {
        float s = 0.909f * v / (1.0f + __expf(-v));   // lipswish
        Cb[(size_t)m * ldcb + nc] = f2bf(s);
      } else if (mode == 2) {
        Cf[(size_t)m * ldc + nc] = v;
      } else {
        Cf[(size_t)m * ldc + nc] = v;
        Cb[(size_t)m * ldcb + nc] = f2bf(v);
      }
    }
  };
  epi(acc00, m0,      n0);
  epi(acc01, m0,      n0 + 16);
  epi(acc10, m0 + 16, n0);
  epi(acc11, m0 + 16, n0 + 16);
}

// ---- f32 -> bf16 weight conversion with zero-padding ---------------------
__global__ void cvt_pad_bf16(const float* __restrict__ src, uint16_t* __restrict__ dst,
                             int cols, int ldd, int total) {
  int i = blockIdx.x * blockDim.x + threadIdx.x;
  if (i >= total) return;
  int r = i / ldd, c = i - r * ldd;
  float v = (c < cols) ? src[(size_t)r * cols + c] : 0.0f;
  dst[i] = f2bf(v);
}

// ---- build z = [t, y, x_knots[s]] (bf16, padded to ZK_) ------------------
__global__ void build_z(uint16_t* __restrict__ zbf, const float* __restrict__ y,
                        const float* __restrict__ coeffs, const float* __restrict__ times,
                        int s) {
  int i = blockIdx.x * blockDim.x + threadIdx.x;
  if (i >= B_ * ZK_) return;
  int b = i / ZK_, c = i - b * ZK_;
  float v;
  if (c == 0)         v = times[s];
  else if (c <= H_)   v = y[(size_t)b * H_ + (c - 1)];
  else if (c < ZRAW_) v = coeffs[((size_t)b * T1_ + s) * C4_ + (c - 1 - H_)];
  else                v = 0.0f;
  zbf[i] = f2bf(v);
}

// ---- x0 = x_knots[0] in bf16 ---------------------------------------------
__global__ void build_x0(uint16_t* __restrict__ x0b, const float* __restrict__ coeffs) {
  int i = blockIdx.x * blockDim.x + threadIdx.x;
  if (i >= B_ * D_) return;
  int b = i / D_, d = i - b * D_;
  x0b[i] = f2bf(coeffs[(size_t)b * T1_ * C4_ + d]);
}

// ---- Euler update: y += drift*dt + jump*levy(u,e); also emit bf16 y ------
// dj is (B, 1024): cols 0..511 = drift pre-scale, 512..1023 = jump pre-scale.
__global__ void update_y(float* __restrict__ y, uint16_t* __restrict__ ybf,
                         const float* __restrict__ dj,
                         const float* __restrict__ U, const float* __restrict__ E,
                         const float* __restrict__ times, int s) {
  int i = blockIdx.x * blockDim.x + threadIdx.x;
  if (i >= B_ * H_) return;
  int b = i / H_, h = i - b * H_;
  float dt = times[s + 1] - times[s];
  float u  = U[(size_t)s * B_ + b];
  float e  = E[(size_t)s * B_ + b];
  // levy(u,e, alpha=1.9, beta=0): const=0, phi=0, S=1
  float num  = __sinf(1.9f * u);
  float den  = __powf(__cosf(u), 0.52631579f);                 // 1/alpha
  float frac = __powf(__cosf(0.9f * u) / e, -0.47368421f);     // (1-alpha)/alpha
  float dL   = num / den * frac;
  float drift = dj[(size_t)b * (2 * H_) + h];
  float jump  = dj[(size_t)b * (2 * H_) + H_ + h];
  float v = y[i] + drift * dt + jump * dL;
  y[i]   = v;
  ybf[i] = f2bf(v);
}

// --------------------------------------------------------------------------
extern "C" void kernel_launch(void* const* d_in, const int* in_sizes, int n_in,
                              void* d_out, int out_size, void* d_ws, size_t ws_size,
                              hipStream_t stream) {
  (void)in_sizes; (void)n_in; (void)out_size; (void)ws_size;
  const float* coeffs  = (const float*)d_in[0];
  const float* times   = (const float*)d_in[1];
  const float* W_in    = (const float*)d_in[2];
  const float* b_in    = (const float*)d_in[3];
  const float* f_W     = (const float*)d_in[4];
  const float* f_b     = (const float*)d_in[5];
  const float* W_noise = (const float*)d_in[6];
  const float* b_noise = (const float*)d_in[7];
  const float* g_W     = (const float*)d_in[8];
  const float* g_b     = (const float*)d_in[9];
  const float* W_init  = (const float*)d_in[10];
  const float* b_init  = (const float*)d_in[11];
  const float* W_dec   = (const float*)d_in[12];
  const float* b_dec   = (const float*)d_in[13];
  const float* U       = (const float*)d_in[14];
  const float* E       = (const float*)d_in[15];
  float* out = (float*)d_out;

  // Workspace carve-out (~20 MB)
  uint8_t* ws = (uint8_t*)d_ws;
  size_t off = 0;
  auto carve = [&](size_t bytes) -> void* {
    void* p = (void*)(ws + off);
    off += (bytes + 255) & ~(size_t)255;
    return p;
  };
  uint16_t* winp = (uint16_t*)carve((size_t)H_ * ZK_ * 2);      // padded W_in  (512x608)
  uint16_t* wnop = (uint16_t*)carve((size_t)H_ * ZK_ * 2);      // padded W_noise
  uint16_t* fwbf = (uint16_t*)carve((size_t)4 * H_ * H_ * 2);   // f MLP weights
  uint16_t* gwbf = (uint16_t*)carve((size_t)4 * H_ * H_ * 2);   // g MLP weights
  uint16_t* wib  = (uint16_t*)carve((size_t)H_ * D_ * 2);       // W_init
  uint16_t* wdb  = (uint16_t*)carve((size_t)OUT_ * H_ * 2);     // W_dec
  uint16_t* zbf  = (uint16_t*)carve((size_t)B_ * ZK_ * 2);
  uint16_t* abuf = (uint16_t*)carve((size_t)B_ * 2 * H_ * 2);   // paired acts (B,1024)
  uint16_t* bbuf = (uint16_t*)carve((size_t)B_ * 2 * H_ * 2);
  uint16_t* x0b  = (uint16_t*)carve((size_t)B_ * D_ * 2);
  uint16_t* ybf  = (uint16_t*)carve((size_t)B_ * H_ * 2);
  float*    yf   = (float*)   carve((size_t)B_ * H_ * 4);
  float*    dj   = (float*)   carve((size_t)B_ * 2 * H_ * 4);   // [drift | jump]

  auto blocks = [](int total) { return (total + 255) / 256; };

  // Weight conversion (every call: no static state allowed)
  { int t = H_ * ZK_;    cvt_pad_bf16<<<blocks(t), 256, 0, stream>>>(W_in,    winp, ZRAW_, ZK_, t); }
  { int t = H_ * ZK_;    cvt_pad_bf16<<<blocks(t), 256, 0, stream>>>(W_noise, wnop, ZRAW_, ZK_, t); }
  { int t = 4 * H_ * H_; cvt_pad_bf16<<<blocks(t), 256, 0, stream>>>(f_W,     fwbf, H_, H_, t); }
  { int t = 4 * H_ * H_; cvt_pad_bf16<<<blocks(t), 256, 0, stream>>>(g_W,     gwbf, H_, H_, t); }
  { int t = H_ * D_;     cvt_pad_bf16<<<blocks(t), 256, 0, stream>>>(W_init,  wib,  D_, D_, t); }
  { int t = OUT_ * H_;   cvt_pad_bf16<<<blocks(t), 256, 0, stream>>>(W_dec,   wdb,  H_, H_, t); }

  auto gemm = [&](const uint16_t* A, int lda,
                  const uint16_t* Wa, const uint16_t* Wb,
                  const float* ba, const float* bb,
                  float* Cf, int ldc, uint16_t* Cb, int ldcb,
                  int N, int K, int mode, int paths, int colOffA, int colOffC) {
    dim3 g(B_ / 64, N / 64, paths);
    gemm_bf16_wmma<<<g, 128, 0, stream>>>(A, lda, Wa, Wb, ba, bb,
                                          Cf, ldc, Cb, ldcb, K, mode, colOffA, colOffC);
  };

  // y0 = x_knots[0] @ W_init^T + b_init  -> yf (f32 state) + ybf (bf16)
  build_x0<<<blocks(B_ * D_), 256, 0, stream>>>(x0b, coeffs);
  gemm(x0b, D_, wib, wib, b_init, b_init, yf, H_, ybf, H_, H_, D_, /*mode=*/3, 1, 0, 0);
  // out[:, 0, :] = y0 @ W_dec^T + b_dec
  gemm(ybf, H_, wdb, wdb, b_dec, b_dec, out, T_ * OUT_, nullptr, 0, OUT_, H_, 2, 1, 0, 0);

  const int HH = H_ * H_;
  for (int s = 0; s < T1_; ++s) {
    build_z<<<blocks(B_ * ZK_), 256, 0, stream>>>(zbf, yf, coeffs, times, s);

    // Depth-paired f/g chain: one launch per depth, blockIdx.z = path.
    // L0: z @ {W_in, W_noise}^T + bias (no activation) -> abuf (B,1024)
    gemm(zbf,  ZK_,    winp,          wnop,          b_in,        b_noise,
         nullptr, 0, abuf, 2 * H_, H_, ZK_, 0, 2, /*colOffA=*/0,   /*colOffC=*/H_);
    // 3 lipswish hidden layers
    gemm(abuf, 2 * H_, fwbf + 0 * HH, gwbf + 0 * HH, f_b + 0 * H_, g_b + 0 * H_,
         nullptr, 0, bbuf, 2 * H_, H_, H_, 1, 2, H_, H_);
    gemm(bbuf, 2 * H_, fwbf + 1 * HH, gwbf + 1 * HH, f_b + 1 * H_, g_b + 1 * H_,
         nullptr, 0, abuf, 2 * H_, H_, H_, 1, 2, H_, H_);
    gemm(abuf, 2 * H_, fwbf + 2 * HH, gwbf + 2 * HH, f_b + 2 * H_, g_b + 2 * H_,
         nullptr, 0, bbuf, 2 * H_, H_, H_, 1, 2, H_, H_);
    // final linear -> dj f32 (B,1024) = [drift | jump]
    gemm(bbuf, 2 * H_, fwbf + 3 * HH, gwbf + 3 * HH, f_b + 3 * H_, g_b + 3 * H_,
         dj, 2 * H_, nullptr, 0, H_, H_, 2, 2, H_, H_);

    update_y<<<blocks(B_ * H_), 256, 0, stream>>>(yf, ybf, dj, U, E, times, s);

    // out[:, s+1, :] = y @ W_dec^T + b_dec   (incremental decode)
    gemm(ybf, H_, wdb, wdb, b_dec, b_dec, out + (size_t)(s + 1) * OUT_, T_ * OUT_,
         nullptr, 0, OUT_, H_, 2, 1, 0, 0);
  }
}
